// ElecInit_14156212207950
// MI455X (gfx1250) — matrix operand order repaired and checked by reading, not compiled, not run on hardware
//
#include <hip/hip_runtime.h>
#include <math.h>

// ---------------- problem constants ----------------
#define BATCH   16384
#define JN      16          // neighbours
#define DIM     256         // feature dim
#define D1C     32
#define D2C     64
#define NEC     8
#define OUTW    1448        // 256 + 512 + 512 + 8 + 128 + 32
#define CUTOFF_F 5.0f

// ---------------- WMMA types / helpers ----------------
typedef __attribute__((ext_vector_type(16))) _Float16 v16h;
typedef __attribute__((ext_vector_type(8)))  float    v8f;
typedef __attribute__((ext_vector_type(4)))  float    f32x4;

union Frag16 { v16h v; f32x4 q[2]; };
union H8     { _Float16 h[8]; f32x4 q; };   // 8 packed f16 = one b128 store

// A-matrix fragment (16x32 f16), source row-major [rows][stride] f16.
// Lane L (<16): M=L, K = 0..7 / 16..23 ; Lane L+16: M=L, K = 8..15 / 24..31.
__device__ __forceinline__ v16h load_afrag(const _Float16* base, int row0,
                                           int stride, int k0) {
    int lane = threadIdx.x & 31;
    int m  = lane & 15;
    int hi = lane >> 4;
    const _Float16* p = base + (size_t)(row0 + m) * stride + k0 + hi * 8;
    Frag16 f;
    f.q[0] = *(const f32x4*)(p);
    f.q[1] = *(const f32x4*)(p + 16);
    return f.v;
}

// B-matrix fragment (32x16 f16) from transposed weights WT row-major [N][K].
// Lane L (<16): N=L, K = 0..7 / 16..23 ; Lane L+16: N=L, K = 8..15 / 24..31.
__device__ __forceinline__ v16h load_bfrag(const _Float16* wt, int n0,
                                           int k0, int K) {
    int lane = threadIdx.x & 31;
    int n  = lane & 15;
    int hi = lane >> 4;
    const _Float16* p = wt + (size_t)(n0 + n) * K + k0 + hi * 8;
    Frag16 f;
    f.q[0] = *(const f32x4*)(p);
    f.q[1] = *(const f32x4*)(p + 16);
    return f.v;
}

__device__ __forceinline__ v8f wmma_f16(v16h a, v16h b, v8f c) {
    return __builtin_amdgcn_wmma_f32_16x16x32_f16(false, a, false, b,
                                                  (short)0, c, false, false);
}

__device__ __forceinline__ float silu_f(float x) {
    return x / (1.0f + expf(-x));
}

// ---------------- weight transpose + f16 convert (K zero-padded) ----------------
// WT[n][kp] = (f16) W[k][n] for k<K, n<N; zero elsewhere. WT is [Nrows][Kpad].
__global__ void convT_kernel(const float* __restrict__ W,
                             _Float16* __restrict__ WT,
                             int K, int Kpad, int N, int Nrows) {
    int idx = blockIdx.x * 256 + threadIdx.x;
    if (idx >= Nrows * Kpad) return;
    int n = idx / Kpad;
    int k = idx - n * Kpad;
    WT[idx] = (n < N && k < K) ? (_Float16)W[(size_t)k * N + n] : (_Float16)0.0f;
}

// ---------------- edge kernel ----------------
// One WG = 16 batch rows x 16 neighbours = 256 edges. 8 waves.
// s_ab holds per edge: phase1 [ h1(32) | env*fcut(8) | zeros(24) ] (K=64 A-matrix),
// then (after barrier) beta(64) reuses the same storage.
__global__ __launch_bounds__(256) void edge_kernel(
    const float* __restrict__ r,           // [B,3]
    const float* __restrict__ R_nb,        // [B,J,3]
    const float* __restrict__ dyn_scales,  // [B,J,8]
    const float* __restrict__ dyn_kernel,  // [B,J,4,32]
    const float* __restrict__ dyn_bias,    // [B,J,32]
    const float* __restrict__ nuc_emb,     // [B,J,256]
    const float* __restrict__ b_f,         // [64]
    const float* __restrict__ W_edge,      // [4,256]
    const _Float16* __restrict__ WfT,      // [64][32]  W_f^T
    const _Float16* __restrict__ WenvT,    // [64][32]  W_env^T (K=8 padded to 32)
    const _Float16* __restrict__ WgT,      // [256][64] W_gamma^T
    float* __restrict__ h_out)             // [B,256]
{
    __shared__ float    s_scaled[256][4];   // 4 KB
    __shared__ _Float16 s_ab[256 * 72];     // 36 KB, stride 72 (bank-safe)

    const int tid = threadIdx.x;
    const int bg0 = blockIdx.x * 16;
    const int lane = tid & 31, wv = tid >> 5, hi = lane >> 4, col = lane & 15;

    // ---- phase 1: per-edge scalar work, fully vectorized loads/stores ----
    {
        int bl = tid >> 4;
        int j  = tid & 15;
        int b  = bg0 + bl;
        int e  = tid;
        float rx = r[b * 3 + 0], ry = r[b * 3 + 1], rz = r[b * 3 + 2];
        const float* Rn = R_nb + ((size_t)b * JN + j) * 3;
        float dx = Rn[0] - rx, dy = Rn[1] - ry, dz = Rn[2] - rz;
        float dist = sqrtf(dx * dx + dy * dy + dz * dz + 1e-12f);
        float invd = 1.0f / dist;
        float lg   = log1pf(dist);
        s_scaled[e][0] = lg;
        s_scaled[e][1] = dx * invd * lg;
        s_scaled[e][2] = dy * invd * lg;
        s_scaled[e][3] = dz * invd * lg;
        float tc   = 1.0f - dist * (1.0f / CUTOFF_F);
        float fcut = (dist < CUTOFF_F) ? tc * tc : 0.0f;

        _Float16* row = s_ab + e * 72;
        const float* dk  = dyn_kernel + ((size_t)b * JN + j) * 4 * D1C;
        const float* dbv = dyn_bias   + ((size_t)b * JN + j) * D1C;
#pragma unroll
        for (int g = 0; g < 4; ++g) {          // 8 h1 columns per group
            H8 pk;
#pragma unroll
            for (int s = 0; s < 2; ++s) {
                int d = g * 8 + s * 4;
                f32x4 k0 = *(const f32x4*)(dk + d);
                f32x4 k1 = *(const f32x4*)(dk + D1C + d);
                f32x4 k2 = *(const f32x4*)(dk + 2 * D1C + d);
                f32x4 k3 = *(const f32x4*)(dk + 3 * D1C + d);
                f32x4 bb = *(const f32x4*)(dbv + d);
#pragma unroll
                for (int i = 0; i < 4; ++i) {
                    float sum = bb[i] + dist * k0[i] + dx * k1[i]
                              + dy * k2[i] + dz * k3[i];
                    pk.h[s * 4 + i] = (_Float16)silu_f(sum);
                }
            }
            *(f32x4*)(row + g * 8) = pk.q;
        }
        // env * f_cut into cols 32..39, zeros in 40..63
        const float* dsc = dyn_scales + ((size_t)b * JN + j) * NEC;
        f32x4 sc0 = *(const f32x4*)(dsc);
        f32x4 sc1 = *(const f32x4*)(dsc + 4);
        H8 pe;
#pragma unroll
        for (int i = 0; i < 8; ++i) {
            float sv = (i < 4) ? sc0[i] : sc1[i - 4];
            float sp = fmaxf(sv, 0.0f) + log1pf(expf(-fabsf(sv)));  // softplus
            pe.h[i] = (_Float16)(expf(-dist * sp) * fcut);
        }
        *(f32x4*)(row + 32) = pe.q;
        H8 z;
#pragma unroll
        for (int i = 0; i < 8; ++i) z.h[i] = (_Float16)0.0f;
        *(f32x4*)(row + 40) = z.q;
        *(f32x4*)(row + 48) = z.q;
        *(f32x4*)(row + 56) = z.q;
    }
    __syncthreads();

    // ---- phase 2: h2 = h1@W_f and env_mix = env@W_env, both as WMMA ----
    float beta_r[8][8];
    {
        int nt  = wv & 3;            // n-tile 0..3 (d2 block)
        int mt0 = (wv >> 2) * 8;     // 8 m-tiles per wave
        int d2c = nt * 16 + col;
        float bfv = b_f[d2c];
        v16h bWf = load_bfrag(WfT,   nt * 16, 0, 32);
        v16h bWe = load_bfrag(WenvT, nt * 16, 0, 32);
#pragma unroll
        for (int mi = 0; mi < 8; ++mi) {
            int mt = mt0 + mi;
            v16h aH = load_afrag(s_ab, mt * 16, 72, 0);   // h1 (K 0..31)
            v16h aE = load_afrag(s_ab, mt * 16, 72, 32);  // env|0 (K 32..63)
            v8f h2 = {}; h2 = wmma_f16(aH, bWf, h2);
            v8f em = {}; em = wmma_f16(aE, bWe, em);
#pragma unroll
            for (int rr = 0; rr < 8; ++rr)
                beta_r[mi][rr] = (h2[rr] + bfv) * em[rr];  // f_cut folded in env
        }
    }
    __syncthreads();        // all reads of h1/env done -> safe to overwrite
    {
        int nt  = wv & 3;
        int mt0 = (wv >> 2) * 8;
        int d2c = nt * 16 + col;
#pragma unroll
        for (int mi = 0; mi < 8; ++mi)
#pragma unroll
            for (int rr = 0; rr < 8; ++rr) {
                int e = (mt0 + mi) * 16 + rr + 8 * hi;
                s_ab[e * 72 + d2c] = (_Float16)beta_r[mi][rr];
            }
    }
    __syncthreads();

    // ---- phase 3: gamma = beta @ W_gamma, fuse edge-embed mult + J-reduce ----
    {
        for (int t = wv * 32; t < wv * 32 + 32; ++t) {
            int bl    = t >> 4;          // batch-local row (= m-tile = one atom)
            int ntile = t & 15;
            int d     = ntile * 16 + col;
            v8f acc = {};
#pragma unroll
            for (int kk = 0; kk < 2; ++kk) {
                v16h a  = load_afrag(s_ab, bl * 16, 72, kk * 32);
                v16h bb = load_bfrag(WgT, ntile * 16, kk * 32, 64);
                acc = wmma_f16(a, bb, acc);
            }
            float we0 = W_edge[d],           we1 = W_edge[DIM + d];
            float we2 = W_edge[2 * DIM + d], we3 = W_edge[3 * DIM + d];
            int bglob = bg0 + bl;
            float partial = 0.0f;
#pragma unroll
            for (int rr = 0; rr < 8; ++rr) {
                int j = rr + 8 * hi;
                int e = bl * 16 + j;
                f32x4 sc = *(const f32x4*)(&s_scaled[e][0]);
                float ev = sc[0] * we0 + sc[1] * we1 + sc[2] * we2 + sc[3] * we3
                         + nuc_emb[((size_t)bglob * JN + j) * DIM + d];
                partial += acc[rr] * ev;
            }
            partial += __shfl_xor(partial, 16, 32);   // fold M 8..15 into 0..7
            if (hi == 0)
                h_out[(size_t)bglob * DIM + d] = partial;
        }
    }
}

// ---------------- MLP kernel ----------------
// One WG = 32 batch rows (2 m-tiles), 4 waves. Every wave processes BOTH
// m-tiles per B-fragment load (halves L2 weight traffic); n-tiles split 4 ways.
__global__ __launch_bounds__(128) void mlp_kernel(
    const float* __restrict__ h_in,     // [B,256]
    const float* __restrict__ ln_g, const float* __restrict__ ln_b,
    const float* __restrict__ b_glu,    // [512]
    const float* __restrict__ b_h,      // [256]
    const float* __restrict__ b_out,    // [1536]
    const float* __restrict__ b_scale,  // [8]
    const float* __restrict__ b_k,      // [128]
    const float* __restrict__ b_b,      // [32]
    const _Float16* __restrict__ WgluT, // [512][256]
    const _Float16* __restrict__ WhT,   // [256][256]
    const _Float16* __restrict__ WoutT, // [1536][256]
    const _Float16* __restrict__ WscaleT, // [16][256] (zero padded)
    const _Float16* __restrict__ WkT,     // [128][256]
    const _Float16* __restrict__ WbT,     // [32][256]
    float* __restrict__ out)            // [B,1448]
{
    __shared__ _Float16 sA[32 * 264];
    __shared__ _Float16 sB[32 * 264];

    const int tid  = threadIdx.x;
    const int wv   = tid >> 5;
    const int lane = tid & 31;
    const int hi   = lane >> 4;
    const int col  = lane & 15;
    const int row0 = blockIdx.x * 32;

    // ---- LayerNorm: wave handles 8 rows; lane owns 8 contiguous cols ----
    for (int rr = 0; rr < 8; ++rr) {
        int rloc = wv * 8 + rr;
        const float* hr = h_in + (size_t)(row0 + rloc) * DIM;
        f32x4 a0 = *(const f32x4*)(hr + lane * 8);
        f32x4 a1 = *(const f32x4*)(hr + lane * 8 + 4);
        float s = a0[0] + a0[1] + a0[2] + a0[3] + a1[0] + a1[1] + a1[2] + a1[3];
#pragma unroll
        for (int off = 16; off > 0; off >>= 1) s += __shfl_xor(s, off, 32);
        float mu = s * (1.0f / 256.0f);
        float var = 0.0f;
#pragma unroll
        for (int i = 0; i < 4; ++i) {
            float d0 = a0[i] - mu; var += d0 * d0;
            float d1 = a1[i] - mu; var += d1 * d1;
        }
#pragma unroll
        for (int off = 16; off > 0; off >>= 1) var += __shfl_xor(var, off, 32);
        float inv = rsqrtf(var * (1.0f / 256.0f) + 1e-6f);
        f32x4 g0 = *(const f32x4*)(ln_g + lane * 8);
        f32x4 g1 = *(const f32x4*)(ln_g + lane * 8 + 4);
        f32x4 bb0 = *(const f32x4*)(ln_b + lane * 8);
        f32x4 bb1 = *(const f32x4*)(ln_b + lane * 8 + 4);
        H8 pk;
#pragma unroll
        for (int i = 0; i < 4; ++i) {
            pk.h[i]     = (_Float16)((a0[i] - mu) * inv * g0[i] + bb0[i]);
            pk.h[4 + i] = (_Float16)((a1[i] - mu) * inv * g1[i] + bb1[i]);
        }
        *(f32x4*)(sA + rloc * 264 + lane * 8) = pk.q;
    }
    __syncthreads();

    // ---- GLU: u = hln @ W_glu ; h = a * sigmoid(g)  (sA -> sB) ----
    for (int nd = wv; nd < 16; nd += 4) {
        v8f aa0 = {}, ag0 = {}, aa1 = {}, ag1 = {};
#pragma unroll
        for (int kk = 0; kk < 8; ++kk) {
            v16h bA = load_bfrag(WgluT, nd * 16,        kk * 32, DIM);
            v16h bG = load_bfrag(WgluT, (nd + 16) * 16, kk * 32, DIM);
            v16h a0 = load_afrag(sA, 0,  264, kk * 32);
            v16h a1 = load_afrag(sA, 16, 264, kk * 32);
            aa0 = wmma_f16(a0, bA, aa0); ag0 = wmma_f16(a0, bG, ag0);
            aa1 = wmma_f16(a1, bA, aa1); ag1 = wmma_f16(a1, bG, ag1);
        }
        int c = nd * 16 + col;
        float ba = b_glu[c], bg = b_glu[DIM + c];
#pragma unroll
        for (int rr = 0; rr < 8; ++rr) {
            float A0 = aa0[rr] + ba, G0 = ag0[rr] + bg;
            sB[(rr + 8 * hi) * 264 + c]      = (_Float16)(A0 / (1.0f + expf(-G0)));
            float A1 = aa1[rr] + ba, G1 = ag1[rr] + bg;
            sB[(16 + rr + 8 * hi) * 264 + c] = (_Float16)(A1 / (1.0f + expf(-G1)));
        }
    }
    __syncthreads();

    // ---- Dense + silu (sB -> sA) ----
    for (int nd = wv; nd < 16; nd += 4) {
        v8f c0 = {}, c1 = {};
#pragma unroll
        for (int kk = 0; kk < 8; ++kk) {
            v16h b = load_bfrag(WhT, nd * 16, kk * 32, DIM);
            c0 = wmma_f16(load_afrag(sB, 0,  264, kk * 32), b, c0);
            c1 = wmma_f16(load_afrag(sB, 16, 264, kk * 32), b, c1);
        }
        int c = nd * 16 + col;
        float bb = b_h[c];
#pragma unroll
        for (int rr = 0; rr < 8; ++rr) {
            sA[(rr + 8 * hi) * 264 + c]      = (_Float16)silu_f(c0[rr] + bb);
            sA[(16 + rr + 8 * hi) * 264 + c] = (_Float16)silu_f(c1[rr] + bb);
        }
    }
    __syncthreads();

    // ---- Out: hout = h @ W_out (sA -> d_out; to_params -> sB) ----
    for (int nd = wv; nd < 96; nd += 4) {
        v8f c0 = {}, c1 = {};
#pragma unroll
        for (int kk = 0; kk < 8; ++kk) {
            v16h b = load_bfrag(WoutT, nd * 16, kk * 32, DIM);
            c0 = wmma_f16(load_afrag(sA, 0,  264, kk * 32), b, c0);
            c1 = wmma_f16(load_afrag(sA, 16, 264, kk * 32), b, c1);
        }
        int c = nd * 16 + col;
        float bo = b_out[c];
#pragma unroll
        for (int rr = 0; rr < 8; ++rr) {
#pragma unroll
            for (int half = 0; half < 2; ++half) {
                int rloc = half * 16 + rr + 8 * hi;
                float v = (half ? c1[rr] : c0[rr]) + bo;
                size_t rg = (size_t)(row0 + rloc);
                if (c < 256)        out[rg * OUTW + c] = v;                      // h_init
                else if (c < 512)   sB[rloc * 264 + (c - 256)] = (_Float16)silu_f(v); // to_params
                else                out[rg * OUTW + (c - 256)] = v;              // h_nb_*
            }
        }
    }
    __syncthreads();

    // ---- heads: scale / kernel / bias from to_params (sB) ----
    for (int t = wv; t < 11; t += 4) {
        const _Float16* W;
        const float* bias;
        int nd_off, outbase, nvalid;
        float mul = 1.0f;
        if (t == 0)      { W = WscaleT; nd_off = 0;     bias = b_scale; outbase = 1280; nvalid = 8;   }
        else if (t < 9)  { W = WkT;     nd_off = t - 1; bias = b_k;     outbase = 1288; nvalid = 128; mul = 0.5f; }
        else             { W = WbT;     nd_off = t - 9; bias = b_b;     outbase = 1416; nvalid = 32;  }
        v8f c0 = {}, c1 = {};
#pragma unroll
        for (int kk = 0; kk < 8; ++kk) {
            v16h b = load_bfrag(W, nd_off * 16, kk * 32, DIM);
            c0 = wmma_f16(load_afrag(sB, 0,  264, kk * 32), b, c0);
            c1 = wmma_f16(load_afrag(sB, 16, 264, kk * 32), b, c1);
        }
        int c = nd_off * 16 + col;
        if (c < nvalid) {
            float bv = bias[c];
#pragma unroll
            for (int rr = 0; rr < 8; ++rr) {
                out[(size_t)(row0 + rr + 8 * hi) * OUTW + outbase + c]      = (c0[rr] + bv) * mul;
                out[(size_t)(row0 + 16 + rr + 8 * hi) * OUTW + outbase + c] = (c1[rr] + bv) * mul;
            }
        }
    }
}

// ---------------- host launch ----------------
static inline size_t align256(size_t x) { return (x + 255) & ~(size_t)255; }

extern "C" void kernel_launch(void* const* d_in, const int* in_sizes, int n_in,
                              void* d_out, int out_size, void* d_ws, size_t ws_size,
                              hipStream_t stream) {
    (void)in_sizes; (void)n_in; (void)out_size; (void)ws_size;

    const float* r          = (const float*)d_in[0];
    const float* R_nb       = (const float*)d_in[1];
    const float* dyn_scales = (const float*)d_in[2];
    const float* dyn_kernel = (const float*)d_in[3];
    const float* dyn_bias   = (const float*)d_in[4];
    const float* nuc_emb    = (const float*)d_in[5];
    const float* W_env      = (const float*)d_in[6];
    const float* W_f        = (const float*)d_in[7];
    const float* b_f        = (const float*)d_in[8];
    const float* W_gamma    = (const float*)d_in[9];
    const float* W_edge     = (const float*)d_in[10];
    const float* ln_g       = (const float*)d_in[11];
    const float* ln_b       = (const float*)d_in[12];
    const float* W_glu      = (const float*)d_in[13];
    const float* b_glu      = (const float*)d_in[14];
    const float* W_h        = (const float*)d_in[15];
    const float* b_h        = (const float*)d_in[16];
    const float* W_out      = (const float*)d_in[17];
    const float* b_out      = (const float*)d_in[18];
    const float* W_scale    = (const float*)d_in[19];
    const float* b_scale    = (const float*)d_in[20];
    const float* W_k        = (const float*)d_in[21];
    const float* b_k        = (const float*)d_in[22];
    const float* W_b        = (const float*)d_in[23];
    const float* b_b        = (const float*)d_in[24];

    float* out = (float*)d_out;

    // workspace layout
    char* ws = (char*)d_ws;
    size_t off = 0;
    float* h_buf = (float*)(ws + off);         off = align256(off + (size_t)BATCH * DIM * 4);
    _Float16* WfT     = (_Float16*)(ws + off); off = align256(off + (size_t)64   * 32  * 2);
    _Float16* WenvT   = (_Float16*)(ws + off); off = align256(off + (size_t)64   * 32  * 2);
    _Float16* WgT     = (_Float16*)(ws + off); off = align256(off + (size_t)256  * 64  * 2);
    _Float16* WgluT   = (_Float16*)(ws + off); off = align256(off + (size_t)512  * 256 * 2);
    _Float16* WhT     = (_Float16*)(ws + off); off = align256(off + (size_t)256  * 256 * 2);
    _Float16* WoutT   = (_Float16*)(ws + off); off = align256(off + (size_t)1536 * 256 * 2);
    _Float16* WscaleT = (_Float16*)(ws + off); off = align256(off + (size_t)16   * 256 * 2);
    _Float16* WkT     = (_Float16*)(ws + off); off = align256(off + (size_t)128  * 256 * 2);
    _Float16* WbT     = (_Float16*)(ws + off); off = align256(off + (size_t)32   * 256 * 2);

    auto blocks = (+[](int elems) { return (elems + 255) / 256; });

    // weight prep: f32 -> f16, transposed [N][Kpad] (K and N zero-padded)
    convT_kernel<<<blocks(64 * 32),    256, 0, stream>>>(W_f,     WfT,     32,  32,  64,   64);
    convT_kernel<<<blocks(64 * 32),    256, 0, stream>>>(W_env,   WenvT,   8,   32,  64,   64);
    convT_kernel<<<blocks(256 * 64),   256, 0, stream>>>(W_gamma, WgT,     64,  64,  256,  256);
    convT_kernel<<<blocks(512 * 256),  256, 0, stream>>>(W_glu,   WgluT,   256, 256, 512,  512);
    convT_kernel<<<blocks(256 * 256),  256, 0, stream>>>(W_h,     WhT,     256, 256, 256,  256);
    convT_kernel<<<blocks(1536 * 256), 256, 0, stream>>>(W_out,   WoutT,   256, 256, 1536, 1536);
    convT_kernel<<<blocks(16 * 256),   256, 0, stream>>>(W_scale, WscaleT, 256, 256, 8,    16);
    convT_kernel<<<blocks(128 * 256),  256, 0, stream>>>(W_k,     WkT,     256, 256, 128,  128);
    convT_kernel<<<blocks(32 * 256),   256, 0, stream>>>(W_b,     WbT,     256, 256, 32,   32);

    // edge stage: B/16 workgroups
    edge_kernel<<<BATCH / 16, 256, 0, stream>>>(
        r, R_nb, dyn_scales, dyn_kernel, dyn_bias, nuc_emb,
        b_f, W_edge, WfT, WenvT, WgT, h_buf);

    // MLP stage: B/32 workgroups
    mlp_kernel<<<BATCH / 32, 128, 0, stream>>>(
        h_buf, ln_g, ln_b, b_glu, b_h, b_out, b_scale, b_k, b_b,
        WgluT, WhT, WoutT, WscaleT, WkT, WbT, out);
}
